// CAFF_3100966388292
// MI455X (gfx1250) — matrix operand order repaired and checked by reading, not compiled, not run on hardware
//
#include <hip/hip_runtime.h>

// ---------------------------------------------------------------------------
// Problem constants
// ---------------------------------------------------------------------------
#define Bn   32
#define Cc   512
#define CIc  256
#define Nn   768
#define HO   256

typedef __bf16 bf16;
typedef __attribute__((ext_vector_type(8)))  __bf16 v8bf;
typedef __attribute__((ext_vector_type(16))) __bf16 v16bf;
typedef __attribute__((ext_vector_type(8)))  float  v8f;

struct P6 { const float* p[6]; };

// ---------------------------------------------------------------------------
// WMMA fragment loaders (CDNA5 16-bit layouts, CDNA5 ISA §7.12.2)
// A (16x32, MxK): lane L<16 -> row M=L, K {0..7, 16..23}; lane L>=16 -> row
// M=L-16, K {8..15, 24..31}.  Two contiguous 16B loads per lane.
// B (32x16, KxN): lane L -> column N=L%16, K same as A (strided scalar loads).
// ---------------------------------------------------------------------------
__device__ __forceinline__ v16bf load_frag_a(const bf16* base, int lda,
                                             int m0, int k0, int lane) {
  int hi = lane >> 4, r = lane & 15;
  const bf16* p = base + (size_t)(m0 + r) * lda + k0 + hi * 8;
  union { v16bf v; v8bf h[2]; } u;
  u.h[0] = *(const v8bf*)(p);
  u.h[1] = *(const v8bf*)(p + 16);
  return u.v;
}

__device__ __forceinline__ v16bf load_frag_b(const bf16* base, int ldb,
                                             int k0, int n0, int lane) {
  int hi = lane >> 4, c = lane & 15;
  const bf16* p = base + (size_t)(k0 + hi * 8) * ldb + n0 + c;
  v16bf v;
  for (int s = 0; s < 8; ++s) {
    v[s] = p[(size_t)s * ldb];
    v[s + 8] = p[(size_t)(s + 16) * ldb];
  }
  return v;
}

// A-fragment from an LDS-resident 16 x Cc row-major tile.
__device__ __forceinline__ v16bf load_frag_a_lds(const bf16* tile, int k0,
                                                 int lane) {
  int hi = lane >> 4, r = lane & 15;
  const bf16* p = tile + r * Cc + k0 + hi * 8;
  union { v16bf v; v8bf h[2]; } u;
  u.h[0] = *(const v8bf*)(p);
  u.h[1] = *(const v8bf*)(p + 16);
  return u.v;
}

__device__ __forceinline__ v8f wmma_bf16(v16bf a, v16bf b, v8f c) {
  return __builtin_amdgcn_wmma_f32_16x16x32_bf16(
      /*neg_a=*/false, a, /*neg_b=*/false, b,
      /*c_mod=*/(short)0, c, /*reuse_a=*/false, /*reuse_b=*/false);
}

// ---------------------------------------------------------------------------
// Kernel 1: weight prep. Converts the 6 projection weights to bf16, computes
// wsum[ci] = (1/C) * sum_co W_w[co][ci] and wb_mean = mean(W_b), and zeroes
// the pooled-attention accumulator (atomicAdd target).
// ---------------------------------------------------------------------------
__global__ void caff_prep_kernel(P6 w, const float* __restrict__ W_w,
                                 const float* __restrict__ W_b,
                                 bf16* __restrict__ wbf,
                                 float* __restrict__ wsum,
                                 float* __restrict__ wb_mean,
                                 float* __restrict__ pooled_att) {
  int gid = blockIdx.x * 256 + threadIdx.x;
  if (blockIdx.x < 3072) {                       // 6 * 256 * 512 = 786432 elems
    int pj  = gid / (CIc * Cc);
    int off = gid % (CIc * Cc);
    wbf[(size_t)pj * CIc * Cc + off] = (bf16)w.p[pj][off];
  } else if (blockIdx.x < 3072 + 96) {           // zero 32*768 pooled slots
    pooled_att[gid - 3072 * 256] = 0.0f;
  } else {                                       // one block: wsum + wb_mean
    int ci = threadIdx.x;                        // 256 threads
    float s = 0.0f;
    for (int co = 0; co < Cc; ++co) s += W_w[(size_t)co * CIc + ci];
    wsum[ci] = s * (1.0f / Cc);
    if (ci == 0) {
      float sb = 0.0f;
      for (int co = 0; co < Cc; ++co) sb += W_b[co];
      *wb_mean = sb * (1.0f / Cc);
    }
  }
}

// ---------------------------------------------------------------------------
// Kernel 2: convert sar/opt to bf16 and compute channel means (B,N).
// grid = 2 streams * 32 batches * 3 chunks of 256 columns.
// ---------------------------------------------------------------------------
__global__ void caff_convert_mean_kernel(const float* __restrict__ sar,
                                         const float* __restrict__ opt,
                                         bf16* __restrict__ sarbf,
                                         bf16* __restrict__ optbf,
                                         float* __restrict__ meanC_sar,
                                         float* __restrict__ meanC_opt) {
  int blk = blockIdx.x;
  int s = blk / (Bn * 3);
  int b = (blk / 3) % Bn;
  int n = (blk % 3) * 256 + threadIdx.x;
  const float* x  = (s ? opt : sar) + (size_t)b * Cc * Nn + n;
  bf16*        xb = (s ? optbf : sarbf) + (size_t)b * Cc * Nn + n;
  float acc = 0.0f;
  for (int c = 0; c < Cc; ++c) {
    float v = x[(size_t)c * Nn];
    xb[(size_t)c * Nn] = (bf16)v;
    acc += v;
  }
  (s ? meanC_opt : meanC_sar)[b * Nn + n] = acc * (1.0f / Cc);
}

// ---------------------------------------------------------------------------
// Kernel 3: all six 1x1-conv projections via WMMA bf16.
// One BLOCK per (pj, batch, m-tile). The 16x512 bf16 weight tile (16KB,
// contiguous in memory) is staged into LDS once per block with
// global_load_async_to_lds_b128 (ASYNCcnt), then each of the 8 waves computes
// 6 n-tiles with 6 accumulators sharing every LDS A-fragment.
// g and theta are stored TRANSPOSED (B,N,CI); phi stays (B,CI,N).
// ---------------------------------------------------------------------------
__global__ void caff_proj_kernel(const bf16* __restrict__ wbf,
                                 const bf16* __restrict__ sarbf,
                                 const bf16* __restrict__ optbf,
                                 P6 bias,
                                 bf16* __restrict__ gx_t, bf16* __restrict__ gy_t,
                                 bf16* __restrict__ thx_t, bf16* __restrict__ thy_t,
                                 bf16* __restrict__ phx,  bf16* __restrict__ phy) {
  __shared__ bf16 tileA[16 * Cc];                // 16 KB

  int mt = blockIdx.x % 16;
  int b  = (blockIdx.x / 16) % Bn;
  int pj = blockIdx.x / (16 * Bn);
  int s = pj / 3, kind = pj % 3;
  int m0 = mt * 16;

  int tid  = threadIdx.x;
  int lane = tid & 31;
  int wave = tid >> 5;

  // ---- async-stage the A (weight) tile: 1024 x 16B chunks, 4 per thread ----
  {
    const bf16* Ag = wbf + (size_t)pj * CIc * Cc + (size_t)m0 * Cc;
    for (int i = 0; i < 4; ++i) {
      int chunk = tid * 4 + i;                   // 0..1023
      const bf16* g = Ag + chunk * 8;            // contiguous region
      unsigned lds = (unsigned)(unsigned long long)(uintptr_t)&tileA[chunk * 8];
      asm volatile("global_load_async_to_lds_b128 %0, %1, off"
                   :: "v"(lds), "v"(g) : "memory");
    }
    asm volatile("s_wait_asynccnt 0x0" ::: "memory");
    __syncthreads();
  }

  const bf16* X = (s ? optbf : sarbf) + (size_t)b * Cc * Nn;

  v8f acc[6] = {};
  for (int k = 0; k < Cc; k += 32) {
    v16bf af = load_frag_a_lds(tileA, k, lane);
    __builtin_prefetch(X + (size_t)(k + 64 + lane) * Nn + wave * 96, 0, 1);
#pragma unroll
    for (int t = 0; t < 6; ++t) {
      int n0 = (wave * 6 + t) * 16;
      v16bf bfg = load_frag_b(X, Nn, k, n0, lane);
      acc[t] = wmma_bf16(af, bfg, acc[t]);
    }
  }

  int hi = lane >> 4, nc = lane & 15;
  const float* bp = bias.p[pj];
  if (kind == 2) {  // phi: (B,CI,N) row-major
    bf16* out = (s ? phy : phx) + (size_t)b * CIc * Nn;
#pragma unroll
    for (int t = 0; t < 6; ++t) {
      int n0 = (wave * 6 + t) * 16;
      for (int r = 0; r < 8; ++r) {
        int ci = m0 + r + 8 * hi;
        out[(size_t)ci * Nn + n0 + nc] = (bf16)(acc[t][r] + bp[ci]);
      }
    }
  } else {          // g / theta: transposed (B,N,CI); 8 contiguous bf16/lane
    bf16* out = (kind == 0 ? (s ? gy_t : gx_t) : (s ? thy_t : thx_t)) +
                (size_t)b * Nn * CIc;
#pragma unroll
    for (int t = 0; t < 6; ++t) {
      int n0 = (wave * 6 + t) * 16;
      union { v8bf v; bf16 e[8]; } st;
      for (int r = 0; r < 8; ++r)
        st.e[r] = (bf16)(acc[t][r] + bp[m0 + r + 8 * hi]);
      *(v8bf*)(out + (size_t)(n0 + nc) * CIc + m0 + 8 * hi) = st.v;
    }
  }
}

// ---------------------------------------------------------------------------
// Kernel 4: attention logits: L[s][b] = theta_t (N x CI) @ phi (CI x N), f32.
// One wave per 16x64 output strip: 4 accumulators share each A-fragment.
// ---------------------------------------------------------------------------
__global__ void caff_logits_kernel(const bf16* __restrict__ thx_t,
                                   const bf16* __restrict__ thy_t,
                                   const bf16* __restrict__ phx,
                                   const bf16* __restrict__ phy,
                                   float* __restrict__ L1,
                                   float* __restrict__ L2) {
  int lane = threadIdx.x & 31;
  int idx  = blockIdx.x * 8 + (threadIdx.x >> 5);
  int jg = idx % 12;                 // group of 4 j-tiles (64 cols)
  int it = (idx / 12) % 48;
  int b  = (idx / 576) % Bn;
  int s  = idx / (576 * Bn);
  int i0 = it * 16, j0 = jg * 64;

  const bf16* A  = (s ? thy_t : thx_t) + (size_t)b * Nn * CIc;
  const bf16* Bp = (s ? phy : phx) + (size_t)b * CIc * Nn;
  float* L = (s ? L2 : L1) + (size_t)b * Nn * Nn;

  v8f acc[4] = {};
  for (int k = 0; k < CIc; k += 32) {
    v16bf af = load_frag_a(A, CIc, i0, k, lane);
#pragma unroll
    for (int t = 0; t < 4; ++t) {
      v16bf bfg = load_frag_b(Bp, Nn, k, j0 + t * 16, lane);
      acc[t] = wmma_bf16(af, bfg, acc[t]);
    }
  }
  int hi = lane >> 4, nc = lane & 15;
#pragma unroll
  for (int t = 0; t < 4; ++t)
    for (int r = 0; r < 8; ++r)
      L[(size_t)(i0 + r + 8 * hi) * Nn + j0 + t * 16 + nc] = acc[t][r];
}

// ---------------------------------------------------------------------------
// Kernel 5: fused dual softmax + elementwise product -> att (bf16, B,N,N).
// One 256-thread block per attention row (3 elems/thread).
// ---------------------------------------------------------------------------
__device__ __forceinline__ float caff_block_red(float v, float* red, bool isMax) {
  int tid = threadIdx.x;
  red[tid] = v;
  __syncthreads();
  for (int off = 128; off > 0; off >>= 1) {
    if (tid < off) {
      float o = red[tid + off];
      red[tid] = isMax ? fmaxf(red[tid], o) : (red[tid] + o);
    }
    __syncthreads();
  }
  float r = red[0];
  __syncthreads();
  return r;
}

__global__ void caff_softmax_fuse_kernel(const float* __restrict__ L1,
                                         const float* __restrict__ L2,
                                         bf16* __restrict__ att) {
  __shared__ float red[256];
  size_t row = blockIdx.x;                       // 0 .. B*N-1
  const float* l1 = L1 + row * Nn;
  const float* l2 = L2 + row * Nn;
  bf16* a = att + row * Nn;
  int tid = threadIdx.x;

  float v1[3], v2[3];
  float m1 = -3.0e38f, m2 = -3.0e38f;
  for (int i = 0; i < 3; ++i) {
    v1[i] = l1[tid + i * 256];
    v2[i] = l2[tid + i * 256];
    m1 = fmaxf(m1, v1[i]);
    m2 = fmaxf(m2, v2[i]);
  }
  m1 = caff_block_red(m1, red, true);
  m2 = caff_block_red(m2, red, true);

  float s1 = 0.0f, s2 = 0.0f;
  for (int i = 0; i < 3; ++i) {
    v1[i] = __expf(v1[i] - m1);
    v2[i] = __expf(v2[i] - m2);
    s1 += v1[i];
    s2 += v2[i];
  }
  s1 = caff_block_red(s1, red, false);
  s2 = caff_block_red(s2, red, false);

  float inv = 1.0f / (s1 * s2);
  for (int i = 0; i < 3; ++i)
    a[tid + i * 256] = (bf16)(v1[i] * v2[i] * inv);
}

// ---------------------------------------------------------------------------
// Kernel 6: y = (att @ g_x) * (att @ g_y), contracted against wsum (the
// collapsed W_w projection + channel mean). One wave per 16x32 strip over
// BOTH streams: 4 accumulators share each A-fragment. Both c-tiles fold into
// a single cross-lane reduction + one atomicAdd per output row.
// ---------------------------------------------------------------------------
__global__ void caff_attn_apply_kernel(const bf16* __restrict__ att,
                                       const bf16* __restrict__ gx_t,
                                       const bf16* __restrict__ gy_t,
                                       const float* __restrict__ wsum,
                                       float* __restrict__ pooled_att) {
  int lane = threadIdx.x & 31;
  int idx  = blockIdx.x * 8 + (threadIdx.x >> 5);
  int cg = idx % 8;                  // group of 2 c-tiles (32 cols)
  int nt = (idx / 8) % 48;
  int b  = idx / 384;
  int n0 = nt * 16, c0 = cg * 32;

  const bf16* A  = att  + (size_t)b * Nn * Nn;
  const bf16* B1 = gx_t + (size_t)b * Nn * CIc;
  const bf16* B2 = gy_t + (size_t)b * Nn * CIc;

  v8f a11 = {}, a12 = {}, a21 = {}, a22 = {};
  for (int k = 0; k < Nn; k += 32) {
    __builtin_prefetch(A + (size_t)(n0 + (lane & 15)) * Nn + k + 64, 0, 1);
    v16bf af  = load_frag_a(A, Nn, n0, k, lane);
    v16bf b1a = load_frag_b(B1, CIc, k, c0, lane);
    v16bf b1b = load_frag_b(B1, CIc, k, c0 + 16, lane);
    v16bf b2a = load_frag_b(B2, CIc, k, c0, lane);
    v16bf b2b = load_frag_b(B2, CIc, k, c0 + 16, lane);
    a11 = wmma_bf16(af, b1a, a11);
    a12 = wmma_bf16(af, b1b, a12);
    a21 = wmma_bf16(af, b2a, a21);
    a22 = wmma_bf16(af, b2b, a22);
  }

  int hi = lane >> 4, nc = lane & 15;
  float wc0 = wsum[c0 + nc];
  float wc1 = wsum[c0 + 16 + nc];
  for (int r = 0; r < 8; ++r) {
    float v = a11[r] * a21[r] * wc0 + a12[r] * a22[r] * wc1;
    v += __shfl_xor(v, 1, 32);
    v += __shfl_xor(v, 2, 32);
    v += __shfl_xor(v, 4, 32);
    v += __shfl_xor(v, 8, 32);
    if (nc == 0)
      atomicAdd(&pooled_att[b * Nn + n0 + r + 8 * hi], v);
  }
}

// ---------------------------------------------------------------------------
// Kernel 7: pooled_final = gamma_att*(pooled_att + wb_mean)
//                        + gamma_opt*meanC_opt + gamma_sar*meanC_sar
// ---------------------------------------------------------------------------
__global__ void caff_finalize_pool_kernel(const float* __restrict__ pooled_att,
                                          const float* __restrict__ wb_mean,
                                          const float* __restrict__ meanC_sar,
                                          const float* __restrict__ meanC_opt,
                                          const float* __restrict__ gsar,
                                          const float* __restrict__ gopt,
                                          const float* __restrict__ gatt,
                                          float* __restrict__ pooled_final) {
  int i = blockIdx.x * 256 + threadIdx.x;        // B*N = 24576
  pooled_final[i] = gatt[0] * (pooled_att[i] + wb_mean[0]) +
                    gopt[0] * meanC_opt[i] + gsar[0] * meanC_sar[i];
}

// ---------------------------------------------------------------------------
// Kernel 8: head — out[b][o] = pooled_final[b] . head_w[o] + head_b[o]
// ---------------------------------------------------------------------------
__global__ void caff_head_kernel(const float* __restrict__ pooled_final,
                                 const float* __restrict__ head_w,
                                 const float* __restrict__ head_b,
                                 float* __restrict__ out) {
  int b = blockIdx.x;
  int o = threadIdx.x;
  const float* hw = head_w + (size_t)o * Nn;
  const float* pf = pooled_final + (size_t)b * Nn;
  float acc = head_b[o];
  for (int n = 0; n < Nn; ++n) acc += pf[n] * hw[n];
  out[b * HO + o] = acc;
}

// ---------------------------------------------------------------------------
// Host-side launcher
// ---------------------------------------------------------------------------
extern "C" void kernel_launch(void* const* d_in, const int* in_sizes, int n_in,
                              void* d_out, int out_size, void* d_ws, size_t ws_size,
                              hipStream_t stream) {
  const float* sar    = (const float*)d_in[0];
  const float* opt    = (const float*)d_in[1];
  const float* W_w    = (const float*)d_in[14];
  const float* W_b    = (const float*)d_in[15];
  const float* head_w = (const float*)d_in[16];
  const float* head_b = (const float*)d_in[17];
  const float* gsar   = (const float*)d_in[18];
  const float* gopt   = (const float*)d_in[19];
  const float* gatt   = (const float*)d_in[20];

  // pj order: 0 g_sar, 1 theta_sar, 2 phi_sar, 3 g_opt, 4 theta_opt, 5 phi_opt
  P6 w, bias;
  w.p[0] = (const float*)d_in[2];  bias.p[0] = (const float*)d_in[3];
  w.p[1] = (const float*)d_in[6];  bias.p[1] = (const float*)d_in[7];
  w.p[2] = (const float*)d_in[10]; bias.p[2] = (const float*)d_in[11];
  w.p[3] = (const float*)d_in[4];  bias.p[3] = (const float*)d_in[5];
  w.p[4] = (const float*)d_in[8];  bias.p[4] = (const float*)d_in[9];
  w.p[5] = (const float*)d_in[12]; bias.p[5] = (const float*)d_in[13];

  // ---- workspace carve-up (256B aligned) ----
  size_t off = 0;
  char* base = (char*)d_ws;
  auto carve = [&](size_t bytes) -> void* {
    void* p = base + off;
    off += (bytes + 255) & ~(size_t)255;
    return p;
  };
  bf16*  sarbf  = (bf16*)carve((size_t)Bn * Cc * Nn * 2);
  bf16*  optbf  = (bf16*)carve((size_t)Bn * Cc * Nn * 2);
  bf16*  wbf    = (bf16*)carve((size_t)6 * CIc * Cc * 2);
  bf16*  gx_t   = (bf16*)carve((size_t)Bn * Nn * CIc * 2);
  bf16*  gy_t   = (bf16*)carve((size_t)Bn * Nn * CIc * 2);
  bf16*  thx_t  = (bf16*)carve((size_t)Bn * Nn * CIc * 2);
  bf16*  thy_t  = (bf16*)carve((size_t)Bn * Nn * CIc * 2);
  bf16*  phx    = (bf16*)carve((size_t)Bn * CIc * Nn * 2);
  bf16*  phy    = (bf16*)carve((size_t)Bn * CIc * Nn * 2);
  float* L1     = (float*)carve((size_t)Bn * Nn * Nn * 4);
  float* L2     = (float*)carve((size_t)Bn * Nn * Nn * 4);
  bf16*  att    = (bf16*)carve((size_t)Bn * Nn * Nn * 2);
  float* mC_sar = (float*)carve((size_t)Bn * Nn * 4);
  float* mC_opt = (float*)carve((size_t)Bn * Nn * 4);
  float* p_att  = (float*)carve((size_t)Bn * Nn * 4);
  float* p_fin  = (float*)carve((size_t)Bn * Nn * 4);
  float* wsum   = (float*)carve(CIc * 4);
  float* wbmean = (float*)carve(256);

  float* out = (float*)d_out;

  caff_prep_kernel<<<3169, 256, 0, stream>>>(w, W_w, W_b, wbf, wsum, wbmean, p_att);
  caff_convert_mean_kernel<<<2 * Bn * 3, 256, 0, stream>>>(
      sar, opt, sarbf, optbf, mC_sar, mC_opt);
  caff_proj_kernel<<<6 * Bn * 16, 256, 0, stream>>>(          // block per (pj,b,mt)
      wbf, sarbf, optbf, bias, gx_t, gy_t, thx_t, thy_t, phx, phy);
  caff_logits_kernel<<<(2 * Bn * 48 * 12) / 8, 256, 0, stream>>>(
      thx_t, thy_t, phx, phy, L1, L2);
  caff_softmax_fuse_kernel<<<Bn * Nn, 256, 0, stream>>>(L1, L2, att);
  caff_attn_apply_kernel<<<(Bn * 48 * 8) / 8, 256, 0, stream>>>(
      att, gx_t, gy_t, wsum, p_att);
  caff_finalize_pool_kernel<<<(Bn * Nn) / 256, 256, 0, stream>>>(
      p_att, wbmean, mC_sar, mC_opt, gsar, gopt, gatt, p_fin);
  caff_head_kernel<<<Bn, HO, 0, stream>>>(p_fin, head_w, head_b, out);
}
